// LRA_89472758710456
// MI455X (gfx1250) — matrix operand order repaired and checked by reading, not compiled, not run on hardware
//
#include <hip/hip_runtime.h>
#include <hip/hip_bf16.h>
#include <math.h>

typedef __attribute__((ext_vector_type(16))) _Float16 v16h;
typedef __attribute__((ext_vector_type(8)))  float    v8f;

#define LAMBD 0.99f

__host__ __device__ constexpr float lpow(int k) {
    float r = 1.0f;
    for (int i = 0; i < k; ++i) r *= LAMBD;
    return r;
}

// -----------------------------------------------------------------------------
// Kernel 1: compute s2_contrib[0..8191] into workspace.
// One workgroup, 32 waves. Wave w handles 256 elements = 16 tiles of 16.
// Local tile scans via v_wmma_f32_16x16x32_f16 (triangular-matmul scan trick),
// tile carries via hierarchical scan (serial-in-lane + shuffle log-scan).
// -----------------------------------------------------------------------------
__global__ __launch_bounds__(1024) void lra_scan_kernel(
    const float* __restrict__ lrs, float* __restrict__ s2c)
{
    __shared__ float tileY[512];
    __shared__ float tileP[512];
    __shared__ float carryY[512];
    __shared__ float carryP[512];

    const int tid  = threadIdx.x;
    const int w    = tid >> 5;    // wave / group id 0..31
    const int lane = tid & 31;
    const int hi   = lane >> 4;
    const int lo   = lane & 15;

    // ---- B matrix (32x16 f16): lane<16 -> column N=lane holds K=e (X data);
    //      lanes 16..31 hold K=16..31 which is zero padding.
    v16h b;
    if (hi == 0) {
        const int base = w * 256 + lo * 16;
        #pragma unroll
        for (int e = 0; e < 16; ++e) {
            float x = lrs[base + e] - lrs[base + e + 1];
            b[e] = (_Float16)x;
        }
    } else {
        #pragma unroll
        for (int e = 0; e < 16; ++e) b[e] = (_Float16)0.0f;
    }

    // ---- A matrices (16x32 f16): lower-triangular Toeplitz lambda^(M-K)
    //      (aT) and all-ones lower triangle (aP); K>=16 columns are zero.
    v16h aT, aP;
    {
        const int M = lo;
        #pragma unroll
        for (int e = 0; e < 16; ++e) {
            const int j = e >> 1, p = e & 1;
            const int K = hi * 16 + e;
            float vt = 0.0f, vp = 0.0f;
            if (K < 16 && K <= M) {
                float pw = 1.0f;
                for (int t = 0; t < (M - K); ++t) pw *= LAMBD;
                vt = pw;
                vp = 1.0f;
            }
            aT[e] = (_Float16)vt;
            aP[e] = (_Float16)vp;
        }
    }

    const v8f zero = {0.0f, 0.0f, 0.0f, 0.0f, 0.0f, 0.0f, 0.0f, 0.0f};
    // dY = local decay-scan, dP = local prefix-sum, for 16 tiles at once.
    v8f dY = __builtin_amdgcn_wmma_f32_16x16x32_f16(
        false, aT, false, b, (short)0, zero, false, false);
    v8f dP = __builtin_amdgcn_wmma_f32_16x16x32_f16(
        false, aP, false, b, (short)0, zero, false, false);

    // Tile-end values: M=15 lives in element 7 of the hi half (lanes 16..31).
    if (hi == 1) {
        tileY[w * 16 + lo] = dY[7];
        tileP[w * 16 + lo] = dP[7];
    }
    __syncthreads();

    // ---- Tile-carry scan over 512 tiles (wave 0 only) ----
    if (w == 0) {
        constexpr float L16  = lpow(16);   // per-tile decay
        constexpr float L256 = lpow(256);  // per-group decay
        const int g = lane;                // each lane scans its group's tiles
        float yAcc = 0.0f, pAcc = 0.0f;
        #pragma unroll 1
        for (int k = 0; k < 16; ++k) {
            const int t = g * 16 + k;
            carryY[t] = yAcc;              // exclusive within group
            carryP[t] = pAcc;
            yAcc = tileY[t] + L16 * yAcc;
            pAcc = tileP[t] + pAcc;
        }
        // inclusive log-scan of the 32 group totals (decay factor L256 for Y)
        float fac = L256;
        #pragma unroll
        for (int d = 1; d < 32; d <<= 1) {
            float yUp = __shfl_up(yAcc, d, 32);
            float pUp = __shfl_up(pAcc, d, 32);
            if (lane >= d) { yAcc += fac * yUp; pAcc += pUp; }
            fac *= fac;
        }
        // exclusive group offsets
        float goY = __shfl_up(yAcc, 1, 32);
        float goP = __shfl_up(pAcc, 1, 32);
        if (g == 0) { goY = 0.0f; goP = 0.0f; }
        float pw = 1.0f;
        #pragma unroll 1
        for (int k = 0; k < 16; ++k) {
            const int t = g * 16 + k;
            carryY[t] += pw * goY;         // lambda^(16k) * group offset
            carryP[t] += goP;
            pw *= L16;
        }
    }
    __syncthreads();

    // ---- Apply carries + closed-form cumsum, write s2_contrib ----
    {
        constexpr float INV1ML = 1.0f / (1.0f - LAMBD);
        const int N    = lo;
        const int tile = w * 16 + N;
        const float cy = carryY[tile];
        const float cp = carryP[tile];
        const int gbase = w * 256 + N * 16;
        float pwv = hi ? lpow(9) : lpow(1);   // lambda^(M+1), M starts at hi*8
        #pragma unroll
        for (int f = 0; f < 8; ++f) {
            const int M = f + hi * 8;
            const float Yg = dY[f] + pwv * cy;
            const float Pg = dP[f] + cp;
            s2c[gbase + M] = (Pg - LAMBD * Yg) * INV1ML;
            pwv *= LAMBD;
        }
    }
}

// -----------------------------------------------------------------------------
// Kernel 2: gather S2 at steps[p] and evaluate the scaling-law formula.
// -----------------------------------------------------------------------------
__global__ void lra_point_kernel(const float* __restrict__ S1,
                                 const int* __restrict__ steps,
                                 const float* __restrict__ L0,
                                 const float* __restrict__ A,
                                 const float* __restrict__ alpha,
                                 const float* __restrict__ C,
                                 const float* __restrict__ s2c,
                                 float* __restrict__ out, int n)
{
    const int p = blockIdx.x * blockDim.x + threadIdx.x;
    if (p >= n) return;
    // S2_all[steps+1] == s2_contrib[steps] (steps in [0, n-2])
    const float s2 = s2c[steps[p]];
    const float s1 = fmaxf(S1[p], 1e-10f);
    out[p] = L0[0] + A[0] * powf(s1, -alpha[0]) - C[0] * s2;
}

extern "C" void kernel_launch(void* const* d_in, const int* in_sizes, int n_in,
                              void* d_out, int out_size, void* d_ws, size_t ws_size,
                              hipStream_t stream)
{
    const float* S1    = (const float*)d_in[0];
    const float* lrs   = (const float*)d_in[1];
    const int*   steps = (const int*)d_in[2];
    const float* L0    = (const float*)d_in[3];
    const float* A     = (const float*)d_in[4];
    const float* alpha = (const float*)d_in[5];
    const float* C     = (const float*)d_in[6];

    float* s2c = (float*)d_ws;            // 8192 floats of scratch
    float* out = (float*)d_out;

    lra_scan_kernel<<<1, 1024, 0, stream>>>(lrs, s2c);

    const int n = out_size;
    lra_point_kernel<<<(n + 255) / 256, 256, 0, stream>>>(
        S1, steps, L0, A, alpha, C, s2c, out, n);
}
